// DualRelationsPropagation_17497696764159
// MI455X (gfx1250) — matrix operand order repaired and checked by reading, not compiled, not run on hardware
//
#include <hip/hip_runtime.h>
#include <hip/hip_bf16.h>
#include <math.h>

typedef __attribute__((ext_vector_type(16))) _Float16 v16h;
typedef __attribute__((ext_vector_type(8)))  _Float16 v8h;
typedef __attribute__((ext_vector_type(8)))  float    v8f;
typedef __attribute__((ext_vector_type(4)))  float    v4f;

#define NH 768
#define NN 256
#define NR 16
#define NL 3
#define NHEADS 8
#define HD 96
#define H2 384
#define RH (NR*NH)

#define GEMM_RELU 1

__device__ __forceinline__ int iclamp(int v, int hi) { return v < hi ? v : hi; }

// registers holding one staged (A,B) sub-tile per thread
struct TileRegs { v4f a0, a1, b0, b1; };

__device__ __forceinline__ void load_tile(
    const float* __restrict__ Ab, long aRowOff, int aCS,
    const float* __restrict__ Bb, long bColOff, int bRS,
    int kA, int kB, TileRegs& t)
{
    if (aCS == 1) {                       // contiguous: 2x global_load_b128
        const v4f* ap = (const v4f*)(Ab + aRowOff + kA);
        t.a0 = ap[0]; t.a1 = ap[1];
    } else {                              // strided: hoistable i*aCS offsets
        const float* ap = Ab + aRowOff + (long)kA * aCS;
#pragma unroll
        for (int i = 0; i < 4; ++i) t.a0[i] = ap[(long)i * aCS];
#pragma unroll
        for (int i = 0; i < 4; ++i) t.a1[i] = ap[(long)(i + 4) * aCS];
    }
    if (bRS == 1) {                       // k-contiguous B: 2x global_load_b128
        const v4f* bp = (const v4f*)(Bb + bColOff + kB);
        t.b0 = bp[0]; t.b1 = bp[1];
    } else {
        const float* bp = Bb + bColOff + (long)kB * bRS;
#pragma unroll
        for (int i = 0; i < 4; ++i) t.b0[i] = bp[(long)i * bRS];
#pragma unroll
        for (int i = 0; i < 4; ++i) t.b1[i] = bp[(long)(i + 4) * bRS];
    }
}

__device__ __forceinline__ void store_tile(_Float16* aDst, _Float16* bDst, const TileRegs& t)
{
    v8h pa, pb;
#pragma unroll
    for (int i = 0; i < 4; ++i) { pa[i] = (_Float16)t.a0[i]; pa[i + 4] = (_Float16)t.a1[i]; }
#pragma unroll
    for (int i = 0; i < 4; ++i) { pb[i] = (_Float16)t.b0[i]; pb[i + 4] = (_Float16)t.b1[i]; }
    *(v8h*)aDst = pa;                     // ds_store_b128
    *(v8h*)bDst = pb;                     // ds_store_b128
}

// ---------------------------------------------------------------------------
// Generic WMMA GEMM: C[z] = A[z] (MxK) * B[z] (KxN) + bias[z], opt. relu.
// Block tile 64x64, 8 waves; each wave a 16x32 C tile = 2 WMMAs sharing an
// A fragment. Double-buffered LDS: one barrier per K-step, next tile's
// global loads issued before the current tile's fragments + WMMAs.
// Requires M%64==0 and K%32==0 (true for every launch here); only the
// B/C column dimension may be ragged (guarded: clamped loads, masked store).
// ---------------------------------------------------------------------------
__global__ __launch_bounds__(256) void gemm_wmma(
    const float* __restrict__ A, long aBase, int aRS, int aCS, long aBat,
    const float* __restrict__ B, long bBase, int bRS, int bCS, long bBat,
    const float* __restrict__ bias, long biasBat,
    float* __restrict__ C, long cBase, int cRS, long cBat,
    int M, int Nn, int K, int flags)
{
    __shared__ __align__(16) _Float16 As[2][64][32];   // row-major MxK tile
    __shared__ __align__(16) _Float16 BsT[2][64][32];  // transposed: [n][k]

    const int z = blockIdx.z;
    const float* Ab = A + aBase + (long)z * aBat;
    const float* Bb = B + bBase + (long)z * bBat;
    float*       Cb = C + cBase + (long)z * cBat;

    const int tid  = threadIdx.x;
    const int lane = tid & 31;
    const int wave = tid >> 5;
    const int wm   = wave >> 1;          // 0..3 : 16-row group
    const int wn   = wave & 1;           // 0..1 : 32-col group
    const int m0   = blockIdx.y * 64;
    const int n0   = blockIdx.x * 64;
    const int m    = lane & 15;
    const int g    = lane >> 4;

    // staging coordinates (fixed per thread)
    const int ar  = tid >> 2;            // A row 0..63
    const int acb = (tid & 3) * 8;       // A k-base 0,8,16,24
    const long aRowOff = (long)(m0 + ar) * aRS;          // always in range
    const int bn  = tid & 63;            // B col 0..63
    const int bkb = (tid >> 6) * 8;      // B k-base 0,8,16,24
    const long bColOff = (long)iclamp(n0 + bn, Nn - 1) * bCS;  // clamped

    v8f acc0 = {}, acc1 = {};

    // prologue: stage tile 0 into buffer 0
    {
        TileRegs t0;
        load_tile(Ab, aRowOff, aCS, Bb, bColOff, bRS, acb, bkb, t0);
        store_tile(&As[0][ar][acb], &BsT[0][bn][bkb], t0);
    }

    int buf = 0;
    for (int k0 = 0; k0 < K; k0 += 32) {
        __syncthreads();
        const bool more = (k0 + 32) < K;

        // issue next tile's global loads before consuming the current tile
        TileRegs nx = {};
        if (more)
            load_tile(Ab, aRowOff, aCS, Bb, bColOff, bRS,
                      k0 + 32 + acb, k0 + 32 + bkb, nx);
        if (k0 + 64 < K) {   // speculative prefetch two tiles ahead
            __builtin_prefetch(Ab + aRowOff + (long)(k0 + 64 + acb) * aCS, 0, 1);
            __builtin_prefetch(Bb + bColOff + (long)(k0 + 64 + bkb) * bRS, 0, 1);
        }

        // ---- fragments via wide LDS loads ----
        const _Float16* Ar = &As[buf][wm * 16 + m][0];
        v8h a0 = *(const v8h*)(Ar + g * 8);
        v8h a1 = *(const v8h*)(Ar + 16 + g * 8);
        v16h af = __builtin_shufflevector(a0, a1, 0,1,2,3,4,5,6,7,8,9,10,11,12,13,14,15);

        const _Float16* Br0 = &BsT[buf][wn * 32 + m][0];
        v8h b00 = *(const v8h*)(Br0 + g * 8);
        v8h b01 = *(const v8h*)(Br0 + 16 + g * 8);
        v16h bf0 = __builtin_shufflevector(b00, b01, 0,1,2,3,4,5,6,7,8,9,10,11,12,13,14,15);
        const _Float16* Br1 = &BsT[buf][wn * 32 + 16 + m][0];
        v8h b10 = *(const v8h*)(Br1 + g * 8);
        v8h b11 = *(const v8h*)(Br1 + 16 + g * 8);
        v16h bf1 = __builtin_shufflevector(b10, b11, 0,1,2,3,4,5,6,7,8,9,10,11,12,13,14,15);

        acc0 = __builtin_amdgcn_wmma_f32_16x16x32_f16(
            false, af, false, bf0, (short)0, acc0, false, false);
        acc1 = __builtin_amdgcn_wmma_f32_16x16x32_f16(
            false, af, false, bf1, (short)0, acc1, false, false);

        // write next tile into the other buffer (synced at top of next iter)
        if (more)
            store_tile(&As[buf ^ 1][ar][acb], &BsT[buf ^ 1][bn][bkb], nx);
        buf ^= 1;
    }

    // ---- store: acc[i] -> row = i + 8*g, col = lane&15 (per 16x16 tile) ----
#pragma unroll
    for (int sub = 0; sub < 2; ++sub) {
        const v8f acc = sub ? acc1 : acc0;
        const int col = n0 + wn * 32 + sub * 16 + m;
        if (col < Nn) {
            float bv = bias ? bias[(long)z * biasBat + col] : 0.0f;
#pragma unroll
            for (int i = 0; i < 8; ++i) {
                int rowc = m0 + wm * 16 + i + 8 * g;   // always < M (M%64==0)
                float v = acc[i] + bv;
                if (flags & GEMM_RELU) v = fmaxf(v, 0.0f);
                Cb[(long)rowc * cRS + col] = v;
            }
        }
    }
}

// ---------------------------------------------------------------------------
// Fused pairwise relations: out[i,j,r] = mask * softmax_r( sum_h
//     relu(fa[i,h] + fb[j,h] + b1[h]) * w2[h,r] + b2[r] )
// One thread per (i,j) pair; w2 (48KB) + b1 staged in LDS.
// ---------------------------------------------------------------------------
__global__ __launch_bounds__(256) void relations_kernel(
    const float* __restrict__ fa, const float* __restrict__ fb,
    const float* __restrict__ b1, const float* __restrict__ w2,
    const float* __restrict__ b2, float* __restrict__ out)
{
    __shared__ float w2s[NH * NR];
    __shared__ float b1s[NH];
    for (int t = threadIdx.x; t < NH * NR; t += 256) w2s[t] = w2[t];
    for (int t = threadIdx.x; t < NH; t += 256) b1s[t] = b1[t];
    __syncthreads();

    const int idx = blockIdx.x * 256 + threadIdx.x;   // 65536 pairs
    const int i = idx >> 8, j = idx & 255;
    const float* far = fa + (long)i * NH;
    const float* fbr = fb + (long)j * NH;

    float acc[NR];
#pragma unroll
    for (int r = 0; r < NR; ++r) acc[r] = 0.0f;

    for (int h = 0; h < NH; ++h) {
        float v = fmaxf(far[h] + fbr[h] + b1s[h], 0.0f);
#pragma unroll
        for (int r = 0; r < NR; ++r) acc[r] = fmaf(v, w2s[h * NR + r], acc[r]);
    }
    float mx = -3.4e38f;
#pragma unroll
    for (int r = 0; r < NR; ++r) { acc[r] += b2[r]; mx = fmaxf(mx, acc[r]); }
    float s = 0.0f;
#pragma unroll
    for (int r = 0; r < NR; ++r) { acc[r] = expf(acc[r] - mx); s += acc[r]; }
    const float inv = (i == j) ? 0.0f : (1.0f / s);
#pragma unroll
    for (int r = 0; r < NR; ++r) out[(long)idx * NR + r] = acc[r] * inv;
}

// ---------------------------------------------------------------------------
__global__ __launch_bounds__(256) void concat_kernel(
    const float* __restrict__ s, const float* __restrict__ q, float* __restrict__ f)
{
    int i = blockIdx.x * 256 + threadIdx.x;
    if (i < NN * NH) f[i] = (i < 64 * NH) ? s[i] : q[i - 64 * NH];
}

__global__ __launch_bounds__(256) void divw_kernel(
    const float* __restrict__ w1, float* __restrict__ waD, float* __restrict__ wbD)
{
    int i = blockIdx.x * 256 + threadIdx.x;
    if (i < NH * NH) {
        float da = w1[i], db = w1[NH * NH + i];
        waD[i] = da + db;
        wbD[i] = db - da;
    }
}

__global__ __launch_bounds__(256) void axpby_kernel(
    const float* __restrict__ a, const float* __restrict__ b,
    float* __restrict__ c, float ca, float cb, int n)
{
    int i = blockIdx.x * 256 + threadIdx.x;
    if (i < n) c[i] = ca * a[i] + cb * b[i];
}

// LayerNorm over rows of 768, optional residual (out may alias R).
__global__ __launch_bounds__(256) void ln_kernel(
    const float* __restrict__ X, const float* __restrict__ R,
    const float* __restrict__ gw, const float* __restrict__ bw,
    float* __restrict__ out)
{
    __shared__ float red[256];
    const int row = blockIdx.x, tid = threadIdx.x;
    float vals[3];
    float s = 0.0f;
#pragma unroll
    for (int c = 0; c < 3; ++c) {
        int col = tid + c * 256;
        float v = X[(long)row * NH + col];
        if (R) v += R[(long)row * NH + col];
        vals[c] = v; s += v;
    }
    red[tid] = s; __syncthreads();
    for (int off = 128; off > 0; off >>= 1) {
        if (tid < off) red[tid] += red[tid + off];
        __syncthreads();
    }
    const float mean = red[0] * (1.0f / NH);
    __syncthreads();
    float vs = 0.0f;
#pragma unroll
    for (int c = 0; c < 3; ++c) { float d = vals[c] - mean; vs += d * d; }
    red[tid] = vs; __syncthreads();
    for (int off = 128; off > 0; off >>= 1) {
        if (tid < off) red[tid] += red[tid + off];
        __syncthreads();
    }
    const float inv = rsqrtf(red[0] * (1.0f / NH) + 1e-5f);
#pragma unroll
    for (int c = 0; c < 3; ++c) {
        int col = tid + c * 256;
        out[(long)row * NH + col] = (vals[c] - mean) * inv * gw[col] + bw[col];
    }
}

// softmax over rows of 256 (attention logits), scale applied first
__global__ __launch_bounds__(256) void softmax_kernel(float* __restrict__ att, float scale)
{
    __shared__ float red[256];
    const int row = blockIdx.x, tid = threadIdx.x;
    float v = att[(long)row * 256 + tid] * scale;
    red[tid] = v; __syncthreads();
    for (int off = 128; off > 0; off >>= 1) {
        if (tid < off) red[tid] = fmaxf(red[tid], red[tid + off]);
        __syncthreads();
    }
    const float mx = red[0]; __syncthreads();
    v = expf(v - mx);
    red[tid] = v; __syncthreads();
    for (int off = 128; off > 0; off >>= 1) {
        if (tid < off) red[tid] += red[tid + off];
        __syncthreads();
    }
    att[(long)row * 256 + tid] = v / red[0];
}

// tiny classifier head: preds[i,c] = b2[c] + sum_h ch[i,h]*w2[h,c]
__global__ __launch_bounds__(256) void cls2_kernel(
    const float* __restrict__ ch, const float* __restrict__ w2,
    const float* __restrict__ b2, float* __restrict__ preds)
{
    int idx = blockIdx.x * 256 + threadIdx.x;
    if (idx < 192 * 3) {
        int i = idx / 3, c = idx % 3;
        float s = b2[c];
        for (int h = 0; h < H2; ++h) s = fmaf(ch[(long)i * H2 + h], w2[h * 3 + c], s);
        preds[idx] = s;
    }
}

// ---------------------------------------------------------------------------
static inline void gemm(hipStream_t st,
    const float* A, long aBase, int aRS, int aCS, long aBat,
    const float* B, long bBase, int bRS, int bCS, long bBat,
    const float* bias, long biasBat,
    float* C, long cBase, int cRS, long cBat,
    int M, int N, int K, int flags, int batch)
{
    dim3 grid((N + 63) / 64, (M + 63) / 64, batch);
    gemm_wmma<<<grid, 256, 0, st>>>(A, aBase, aRS, aCS, aBat,
                                    B, bBase, bRS, bCS, bBat,
                                    bias, biasBat,
                                    C, cBase, cRS, cBat, M, N, K, flags);
}

extern "C" void kernel_launch(void* const* d_in, const int* in_sizes, int n_in,
                              void* d_out, int out_size, void* d_ws, size_t ws_size,
                              hipStream_t stream)
{
    const float* sf      = (const float*)d_in[0];
    const float* qf      = (const float*)d_in[1];
    const float* sim_w1  = (const float*)d_in[3];
    const float* sim_b1  = (const float*)d_in[4];
    const float* sim_w2  = (const float*)d_in[5];
    const float* sim_b2  = (const float*)d_in[6];
    const float* div_w1  = (const float*)d_in[7];
    const float* div_b1  = (const float*)d_in[8];
    const float* div_w2  = (const float*)d_in[9];
    const float* div_b2  = (const float*)d_in[10];
    const float* dis_w1  = (const float*)d_in[11];
    const float* dis_b1  = (const float*)d_in[12];
    const float* dis_w2  = (const float*)d_in[13];
    const float* dis_b2  = (const float*)d_in[14];
    const float* dis_ln_g= (const float*)d_in[15];
    const float* dis_ln_b= (const float*)d_in[16];
    const float* rel_W   = (const float*)d_in[17];
    const float* rel_b   = (const float*)d_in[18];
    const float* agg_w1  = (const float*)d_in[19];
    const float* agg_b1  = (const float*)d_in[20];
    const float* agg_w2  = (const float*)d_in[21];
    const float* agg_b2  = (const float*)d_in[22];
    const float* mha_wq  = (const float*)d_in[23];
    const float* mha_bq  = (const float*)d_in[24];
    const float* mha_wk  = (const float*)d_in[25];
    const float* mha_bk  = (const float*)d_in[26];
    const float* mha_wv  = (const float*)d_in[27];
    const float* mha_bv  = (const float*)d_in[28];
    const float* mha_wo  = (const float*)d_in[29];
    const float* mha_bo  = (const float*)d_in[30];
    const float* ln_g    = (const float*)d_in[31];
    const float* ln_b    = (const float*)d_in[32];
    const float* cls_w1  = (const float*)d_in[33];
    const float* cls_b1  = (const float*)d_in[34];
    const float* cls_w2  = (const float*)d_in[35];
    const float* cls_b2  = (const float*)d_in[36];

    // output layout: preds(576) | x(196608) | sim(1048576) | div(1048576)
    float* out   = (float*)d_out;
    float* preds = out;
    float* x     = out + 576;
    float* simO  = x + NN * NH;
    float* divO  = simO + NN * NN * NR;

    // workspace
    float* ws   = (float*)d_ws;
    float* f    = ws;                 ws += NN * NH;
    float* fa   = ws;                 ws += NN * NH;
    float* fb   = ws;                 ws += NN * NH;
    float* waD  = ws;                 ws += NH * NH;
    float* wbD  = ws;                 ws += NH * NH;
    float* comb = ws;                 ws += NN * NN * NR;
    float* h1   = ws;                 ws += NN * H2;
    float* xpre = ws;                 ws += NN * NH;
    float* tbuf = ws;                 ws += NR * NN * NH;
    float* agg  = ws;                 ws += NN * RH;
    float* a1   = ws;                 ws += NN * NH;
    float* av   = ws;                 ws += NN * NH;
    float* qb   = ws;                 ws += NN * NH;
    float* kb   = ws;                 ws += NN * NH;
    float* vb   = ws;                 ws += NN * NH;
    float* ob   = ws;                 ws += NN * NH;
    float* att  = ws;                 ws += NHEADS * NN * NN;
    float* o2   = ws;                 ws += NN * NH;
    float* chid = ws;                 ws += 192 * H2;
    (void)ws_size; (void)in_sizes; (void)n_in; (void)out_size;

    // f = concat(support, query)
    concat_kernel<<<(NN * NH + 255) / 256, 256, 0, stream>>>(sf, qf, f);

    // ---- sim relations ----
    gemm(stream, f, 0, NH, 1, 0, sim_w1, 0, NH, 1, 0, nullptr, 0,
         fa, 0, NH, 0, NN, NH, NH, 0, 1);
    gemm(stream, f, 0, NH, 1, 0, sim_w1, (long)NH * NH, NH, 1, 0, nullptr, 0,
         fb, 0, NH, 0, NN, NH, NH, 0, 1);
    relations_kernel<<<256, 256, 0, stream>>>(fa, fb, sim_b1, sim_w2, sim_b2, simO);

    // ---- div relations (wa' = da+db, wb' = db-da) ----
    divw_kernel<<<(NH * NH + 255) / 256, 256, 0, stream>>>(div_w1, waD, wbD);
    gemm(stream, f, 0, NH, 1, 0, waD, 0, NH, 1, 0, nullptr, 0,
         fa, 0, NH, 0, NN, NH, NH, 0, 1);
    gemm(stream, f, 0, NH, 1, 0, wbD, 0, NH, 1, 0, nullptr, 0,
         fb, 0, NH, 0, NN, NH, NH, 0, 1);
    relations_kernel<<<256, 256, 0, stream>>>(fa, fb, div_b1, div_w2, div_b2, divO);

    // comb = 0.6*sim + 0.4*div
    axpby_kernel<<<(NN * NN * NR + 255) / 256, 256, 0, stream>>>(
        simO, divO, comb, 0.6f, 0.4f, NN * NN * NR);

    // ---- distance encoder: x = LN(relu(f@W1+b1)@W2+b2) ----
    gemm(stream, f, 0, NH, 1, 0, dis_w1, 0, H2, 1, 0, dis_b1, 0,
         h1, 0, H2, 0, NN, H2, NH, GEMM_RELU, 1);
    gemm(stream, h1, 0, H2, 1, 0, dis_w2, 0, NH, 1, 0, dis_b2, 0,
         xpre, 0, NH, 0, NN, NH, H2, 0, 1);
    ln_kernel<<<NN, 256, 0, stream>>>(xpre, nullptr, dis_ln_g, dis_ln_b, x);

    const float scale = 1.0f / sqrtf((float)HD);

    for (int l = 0; l < NL; ++l) {
        // t[r] = x @ rel_W[l,r] + rel_b[l,r]   (batched over r=16)
        gemm(stream, x, 0, NH, 1, 0,
             rel_W, (long)l * NR * NH * NH, NH, 1, (long)NH * NH,
             rel_b + (long)l * NR * NH, NH,
             tbuf, 0, NH, (long)NN * NH, NN, NH, NH, 0, NR);

        // agg[:,r,:] = comb[:,:,r] @ t[r]   (batched over r; strided A)
        gemm(stream, comb, 0, NN * NR, NR, 1,
             tbuf, 0, NH, 1, (long)NN * NH,
             nullptr, 0,
             agg, 0, RH, NH, NN, NH, NN, 0, NR);

        // a = relu(agg @ agg_w1 + b1) @ agg_w2 + b2
        gemm(stream, agg, 0, RH, 1, 0,
             agg_w1, (long)l * RH * NH, NH, 1, 0, agg_b1 + (long)l * NH, 0,
             a1, 0, NH, 0, NN, NH, RH, GEMM_RELU, 1);
        gemm(stream, a1, 0, NH, 1, 0,
             agg_w2, (long)l * NH * NH, NH, 1, 0, agg_b2 + (long)l * NH, 0,
             av, 0, NH, 0, NN, NH, NH, 0, 1);

        // q from a; k,v from x
        gemm(stream, av, 0, NH, 1, 0, mha_wq, (long)l * NH * NH, NH, 1, 0,
             mha_bq + (long)l * NH, 0, qb, 0, NH, 0, NN, NH, NH, 0, 1);
        gemm(stream, x, 0, NH, 1, 0, mha_wk, (long)l * NH * NH, NH, 1, 0,
             mha_bk + (long)l * NH, 0, kb, 0, NH, 0, NN, NH, NH, 0, 1);
        gemm(stream, x, 0, NH, 1, 0, mha_wv, (long)l * NH * NH, NH, 1, 0,
             mha_bv + (long)l * NH, 0, vb, 0, NH, 0, NN, NH, NH, 0, 1);

        // att[h] = q_h @ k_h^T (batched over heads; B transposed via strides)
        gemm(stream, qb, 0, NH, 1, HD,
             kb, 0, 1, NH, HD,
             nullptr, 0,
             att, 0, NN, (long)NN * NN, NN, NN, HD, 0, NHEADS);
        softmax_kernel<<<NHEADS * NN, 256, 0, stream>>>(att, scale);

        // o[:,h*96:] = att[h] @ v_h
        gemm(stream, att, 0, NN, 1, (long)NN * NN,
             vb, 0, NH, 1, HD,
             nullptr, 0,
             ob, 0, NH, HD, NN, HD, NN, 0, NHEADS);

        // o2 = o @ wo + bo ; x = LN(o2 + x)
        gemm(stream, ob, 0, NH, 1, 0, mha_wo, (long)l * NH * NH, NH, 1, 0,
             mha_bo + (long)l * NH, 0, o2, 0, NH, 0, NN, NH, NH, 0, 1);
        ln_kernel<<<NN, 256, 0, stream>>>(o2, x, ln_g + (long)l * NH,
                                          ln_b + (long)l * NH, x);
    }

    // classifier on query rows (x[64:])
    gemm(stream, x, (long)64 * NH, NH, 1, 0,
         cls_w1, 0, H2, 1, 0, cls_b1, 0,
         chid, 0, H2, 0, 192, H2, NH, GEMM_RELU, 1);
    cls2_kernel<<<3, 256, 0, stream>>>(chid, cls_w2, cls_b2, preds);
}
